// Attention_47605417508937
// MI455X (gfx1250) — matrix operand order repaired and checked by reading, not compiled, not run on hardware
//
#include <hip/hip_runtime.h>
#include <hip/hip_bf16.h>

// ---------------------------------------------------------------------------
// Attention pipeline for MI455X (gfx1250).
//   Stage 0: one-shot fp32->bf16 conversion of activations + weights.
//   All matmuls: v_wmma_f32_16x16x32_bf16 (f32 accumulate), staged by the
//   Tensor Data Mover (tensor_load_to_lds) with hardware LDS row padding
//   (1 dword per 16 dwords -> 17-dword rows, bank-conflict free), NT GEMM double
//   buffered, synced with s_wait_tensorcnt + workgroup barriers.
// ---------------------------------------------------------------------------

typedef __attribute__((ext_vector_type(16))) __bf16 v16bf;
typedef __attribute__((ext_vector_type(8)))  float  v8f;
typedef __attribute__((ext_vector_type(4)))  unsigned v4u;
typedef __attribute__((ext_vector_type(8)))  int      v8i_t;
typedef __attribute__((ext_vector_type(4)))  int      v4i_t;

#define BM 128
#define BN 128
#define BK 32
#define ROWDW 17   // 16 data dwords (32 bf16) + 1 pad dword (TDM pad reproduces this)

#if __has_builtin(__builtin_amdgcn_tensor_load_to_lds) && \
    __has_builtin(__builtin_amdgcn_s_wait_tensorcnt)
#define HAS_TDM 1
#else
#define HAS_TDM 0
#endif

union FragAB { unsigned u[8]; v16bf v; };

__device__ __forceinline__ unsigned short f2bf(float f) {
  unsigned u = __builtin_bit_cast(unsigned, f);
  u += 0x7fffu + ((u >> 16) & 1u);            // round-to-nearest-even
  return (unsigned short)(u >> 16);
}

#if __has_builtin(__builtin_amdgcn_cvt_pk_bf16_f32)
typedef __attribute__((ext_vector_type(2))) __bf16 v2bf;
__device__ __forceinline__ unsigned pack2(float lo, float hi) {
  v2bf r = __builtin_amdgcn_cvt_pk_bf16_f32(lo, hi);   // v_cvt_pk_bf16_f32
  return __builtin_bit_cast(unsigned, r);
}
#else
__device__ __forceinline__ unsigned pack2(float lo, float hi) {
  return (unsigned)f2bf(lo) | ((unsigned)f2bf(hi) << 16);
}
#endif

#if HAS_TDM
// Issue one TDM descriptor: 2-D bf16 tile [tile_d1 rows x tile_d0 elems],
// source row stride = ld elements, dest LDS padded +1 dword per 16 dwords.
// Must be executed by a single wave (EXEC is ignored by the TDM).
__device__ __forceinline__ void tdm_load_tile_bf16(unsigned lds_addr,
                                                   const unsigned short* gsrc,
                                                   unsigned tile_d0, unsigned tile_d1,
                                                   unsigned ld, unsigned rows)
{
  unsigned long long ga = (unsigned long long)(size_t)gsrc;
  v4u g0;
  g0[0] = 1u;                                            // count=1, no gather
  g0[1] = lds_addr;                                      // lds_addr [63:32]
  g0[2] = (unsigned)(ga & 0xffffffffu);                  // global_addr lo
  g0[3] = (unsigned)((ga >> 32) & 0x01ffffffu) | (2u << 30);  // addr hi | type=2

  v8i_t g1;
  // data_size=2B (code 1), pad_enable=1, pad_interval=code 3 (16 DWORDs),
  // pad_amount=code 0 (1 DWORD), workgroup_mask=0.
  g1[0] = (int)((1u << 16) | (1u << 20) | (3u << 22));
  g1[1] = (int)((ld & 0xffffu) << 16);                   // tensor_dim0 lo (abar=0)
  g1[2] = (int)(((ld >> 16) & 0xffffu) | ((rows & 0xffffu) << 16)); // d0 hi | d1 lo
  g1[3] = (int)(((rows >> 16) & 0xffffu) | (tile_d0 << 16));        // d1 hi | tile_dim0
  g1[4] = (int)(tile_d1 & 0xffffu);                      // tile_dim1 | tile_dim2=0
  g1[5] = (int)ld;                                       // tensor_dim0_stride lo
  g1[6] = 0;                                             // stride hi | dim1_stride lo
  g1[7] = 0;

  v4i_t z4 = {0, 0, 0, 0};
  v8i_t z8 = {0, 0, 0, 0, 0, 0, 0, 0};
  __builtin_amdgcn_tensor_load_to_lds(g0, g1, z4, z4, z8, 0);
}
#endif

// Manual bf16 tile staging (fallback), 17-dword padded rows.
__device__ __forceinline__ void stage_tile_bf16(unsigned* slds, const unsigned short* g,
                                                int row0, int k0, int ld, int rows, int tid)
{
#pragma unroll
  for (int c = tid; c < rows * 4; c += 256) {
    int r = c >> 2, seg = c & 3;
    uint4 u = *(const uint4*)(g + (size_t)(row0 + r) * ld + k0 + seg * 8);
    slds[r * ROWDW + seg * 4 + 0] = u.x;
    slds[r * ROWDW + seg * 4 + 1] = u.y;
    slds[r * ROWDW + seg * 4 + 2] = u.z;
    slds[r * ROWDW + seg * 4 + 3] = u.w;
  }
}

// ---------------------------------------------------------------------------
// Kernel 0: one-shot fp32 -> bf16 (n must be a multiple of 4).
// ---------------------------------------------------------------------------
__global__ __launch_bounds__(256)
void f32_to_bf16_kernel(const float* __restrict__ in, unsigned short* __restrict__ out,
                        int n)
{
  int i = (blockIdx.x * 256 + threadIdx.x) * 4;
  if (i < n) {
    float4 f = *(const float4*)(in + i);
    uint2 r;
    r.x = pack2(f.x, f.y);
    r.y = pack2(f.z, f.w);
    *(uint2*)(out + i) = r;
  }
}

// ---------------------------------------------------------------------------
// Kernel 1: NT GEMM bf16 x bf16, C[m][n] = sum_k A[m][k]*B[n][k] (+bias).
// TDM-staged, double-buffered LDS; wave 0 drives the Tensor Data Mover.
// BF16OUT=true : Ch[m][n] = bf16(acc + bias[n])   (projections)
// BF16OUT=false: Cf[m][n] = acc                   (scores)
// ---------------------------------------------------------------------------
template <bool BF16OUT>
__global__ __launch_bounds__(256)
void gemm_nt_bf16_kernel(const unsigned short* __restrict__ A,
                         const unsigned short* __restrict__ B,
                         const float* __restrict__ bias,
                         float* __restrict__ Cf, unsigned short* __restrict__ Ch,
                         int M, int N, int K)
{
  __shared__ unsigned sA[2][BM * ROWDW];
  __shared__ unsigned sB[2][BN * ROWDW];

  const int tid   = threadIdx.x;
  const int lane  = tid & 31;
  const int wave  = tid >> 5;
  const int waveM = wave >> 1;
  const int waveN = wave & 1;
  const int half  = lane >> 4;
  const int l16   = lane & 15;
  const int m_blk = blockIdx.y * BM;
  const int n_blk = blockIdx.x * BN;

  v8f acc[2][4];
#pragma unroll
  for (int tm = 0; tm < 2; ++tm)
#pragma unroll
    for (int tn = 0; tn < 4; ++tn)
#pragma unroll
      for (int i = 0; i < 8; ++i) acc[tm][tn][i] = 0.0f;

#if HAS_TDM
  if (wave == 0) {
    tdm_load_tile_bf16((unsigned)(size_t)(void*)&sA[0][0],
                       A + (size_t)m_blk * K, BK, BM, (unsigned)K, (unsigned)BM);
    tdm_load_tile_bf16((unsigned)(size_t)(void*)&sB[0][0],
                       B + (size_t)n_blk * K, BK, BN, (unsigned)K, (unsigned)BN);
  }
  int buf = 0;
  for (int k0 = 0; k0 < K; k0 += BK) {
    if (wave == 0) __builtin_amdgcn_s_wait_tensorcnt(0);
    __syncthreads();                         // tiles[buf] ready; buf^1 free
    int nk = k0 + BK;
    if (wave == 0 && nk < K) {
      tdm_load_tile_bf16((unsigned)(size_t)(void*)&sA[buf ^ 1][0],
                         A + (size_t)m_blk * K + nk, BK, BM, (unsigned)K, (unsigned)BM);
      tdm_load_tile_bf16((unsigned)(size_t)(void*)&sB[buf ^ 1][0],
                         B + (size_t)n_blk * K + nk, BK, BN, (unsigned)K, (unsigned)BN);
    }
#else
  int buf = 0;
  for (int k0 = 0; k0 < K; k0 += BK) {
    __syncthreads();
    stage_tile_bf16(sA[0], A, m_blk, k0, K, BM, tid);
    stage_tile_bf16(sB[0], B, n_blk, k0, K, BN, tid);
    __syncthreads();
#endif

    FragAB a[2], b[4];
#pragma unroll
    for (int tm = 0; tm < 2; ++tm) {
      int row = waveM * 32 + tm * 16 + l16;
#pragma unroll
      for (int j = 0; j < 8; ++j) {
        int dw = half * 8 + j;
        a[tm].u[j] = sA[buf][row * ROWDW + dw];
      }
    }
#pragma unroll
    for (int tn = 0; tn < 4; ++tn) {
      int row = waveN * 64 + tn * 16 + l16;
#pragma unroll
      for (int j = 0; j < 8; ++j)
        b[tn].u[j] = sB[buf][row * ROWDW + half * 8 + j];
    }
#pragma unroll
    for (int tm = 0; tm < 2; ++tm)
#pragma unroll
      for (int tn = 0; tn < 4; ++tn)
        acc[tm][tn] = __builtin_amdgcn_wmma_f32_16x16x32_bf16(
            false, a[tm].v, false, b[tn].v, (short)0, acc[tm][tn], false, false);
#if HAS_TDM
    buf ^= 1;
#endif
  }

#pragma unroll
  for (int tm = 0; tm < 2; ++tm) {
    int mBase = m_blk + waveM * 32 + tm * 16 + half * 8;
#pragma unroll
    for (int tn = 0; tn < 4; ++tn) {
      int n = n_blk + waveN * 64 + tn * 16 + l16;
      if (BF16OUT) {
        float bv = bias[n];
#pragma unroll
        for (int i = 0; i < 8; ++i)
          Ch[(size_t)(mBase + i) * N + n] = f2bf(acc[tm][tn][i] + bv);
      } else {
#pragma unroll
        for (int i = 0; i < 8; ++i)
          Cf[(size_t)(mBase + i) * N + n] = acc[tm][tn][i];
      }
    }
  }
}

// ---------------------------------------------------------------------------
// Kernel 2: row softmax. One 256-thread block per 4096-wide fp32 row -> bf16.
// ---------------------------------------------------------------------------
__global__ __launch_bounds__(256)
void softmax_rows_kernel(const float* __restrict__ S, unsigned short* __restrict__ P,
                         int cols)
{
  __shared__ float red[8];
  const int row = blockIdx.x;
  const int tid = threadIdx.x;
  const float* src = S + (size_t)row * cols;
  unsigned short* dst = P + (size_t)row * cols;

  float v[16];
#pragma unroll
  for (int i = 0; i < 16; ++i) v[i] = src[tid + i * 256];

  float mx = -3.4e38f;
#pragma unroll
  for (int i = 0; i < 16; ++i) mx = fmaxf(mx, v[i]);
#pragma unroll
  for (int off = 16; off >= 1; off >>= 1) mx = fmaxf(mx, __shfl_xor(mx, off, 32));
  if ((tid & 31) == 0) red[tid >> 5] = mx;
  __syncthreads();
  float rmx = red[0];
#pragma unroll
  for (int w = 1; w < 8; ++w) rmx = fmaxf(rmx, red[w]);

  float s = 0.0f;
#pragma unroll
  for (int i = 0; i < 16; ++i) { v[i] = __expf(v[i] - rmx); s += v[i]; }
#pragma unroll
  for (int off = 16; off >= 1; off >>= 1) s += __shfl_xor(s, off, 32);
  __syncthreads();
  if ((tid & 31) == 0) red[tid >> 5] = s;
  __syncthreads();
  float rs = 0.0f;
#pragma unroll
  for (int w = 0; w < 8; ++w) rs += red[w];
  float inv = 1.0f / rs;

#pragma unroll
  for (int i = 0; i < 16; ++i) dst[tid + i * 256] = f2bf(v[i] * inv);
}

// ---------------------------------------------------------------------------
// Kernel 3: NN GEMM bf16 -> f32 (out = P @ Vb).
// A tile via TDM; B tile transposed into LDS by the waves (16-bit scatter).
// ---------------------------------------------------------------------------
__global__ __launch_bounds__(256)
void gemm_nn_bf16_kernel(const unsigned short* __restrict__ A,
                         const unsigned short* __restrict__ B,
                         float* __restrict__ C, int M, int N, int K)
{
  __shared__ unsigned sA[BM * ROWDW];
  __shared__ unsigned sB[BN * ROWDW];
  unsigned short* sBh = (unsigned short*)sB;   // [n][k], stride 2*ROWDW bf16

  const int tid   = threadIdx.x;
  const int lane  = tid & 31;
  const int wave  = tid >> 5;
  const int waveM = wave >> 1;
  const int waveN = wave & 1;
  const int half  = lane >> 4;
  const int l16   = lane & 15;
  const int m_blk = blockIdx.y * BM;
  const int n_blk = blockIdx.x * BN;

  v8f acc[2][4];
#pragma unroll
  for (int tm = 0; tm < 2; ++tm)
#pragma unroll
    for (int tn = 0; tn < 4; ++tn)
#pragma unroll
      for (int i = 0; i < 8; ++i) acc[tm][tn][i] = 0.0f;

  for (int k0 = 0; k0 < K; k0 += BK) {
    __syncthreads();                    // previous tiles fully consumed
#if HAS_TDM
    if (wave == 0)
      tdm_load_tile_bf16((unsigned)(size_t)(void*)&sA[0],
                         A + (size_t)m_blk * K + k0, BK, BM, (unsigned)K, (unsigned)BM);
#else
    stage_tile_bf16(sA, A, m_blk, k0, K, BM, tid);
#endif
    // B tile: coalesced row reads of V, 16-bit transposed scatter into LDS.
#pragma unroll
    for (int c = tid; c < BK * 16; c += 256) {   // 32 k-rows x 16 segments
      int kr = c >> 4, seg = c & 15;
      union { uint4 q; unsigned short h[8]; } t;
      t.q = *(const uint4*)(B + (size_t)(k0 + kr) * N + n_blk + seg * 8);
#pragma unroll
      for (int e = 0; e < 8; ++e)
        sBh[(seg * 8 + e) * (2 * ROWDW) + kr] = t.h[e];
    }
#if HAS_TDM
    if (wave == 0) __builtin_amdgcn_s_wait_tensorcnt(0);
#endif
    __syncthreads();

    FragAB a[2], b[4];
#pragma unroll
    for (int tm = 0; tm < 2; ++tm) {
      int row = waveM * 32 + tm * 16 + l16;
#pragma unroll
      for (int j = 0; j < 8; ++j) {
        int dw = half * 8 + j;
        a[tm].u[j] = sA[row * ROWDW + dw];
      }
    }
#pragma unroll
    for (int tn = 0; tn < 4; ++tn) {
      int row = waveN * 64 + tn * 16 + l16;
#pragma unroll
      for (int j = 0; j < 8; ++j)
        b[tn].u[j] = sB[row * ROWDW + half * 8 + j];
    }
#pragma unroll
    for (int tm = 0; tm < 2; ++tm)
#pragma unroll
      for (int tn = 0; tn < 4; ++tn)
        acc[tm][tn] = __builtin_amdgcn_wmma_f32_16x16x32_bf16(
            false, a[tm].v, false, b[tn].v, (short)0, acc[tm][tn], false, false);
  }

#pragma unroll
  for (int tm = 0; tm < 2; ++tm) {
    int mBase = m_blk + waveM * 32 + tm * 16 + half * 8;
#pragma unroll
    for (int tn = 0; tn < 4; ++tn) {
      int n = n_blk + waveN * 64 + tn * 16 + l16;
#pragma unroll
      for (int i = 0; i < 8; ++i)
        C[(size_t)(mBase + i) * N + n] = acc[tm][tn][i];
    }
  }
}

// ---------------------------------------------------------------------------
extern "C" void kernel_launch(void* const* d_in, const int* in_sizes, int n_in,
                              void* d_out, int out_size, void* d_ws, size_t ws_size,
                              hipStream_t stream) {
  (void)in_sizes; (void)n_in; (void)out_size; (void)ws_size;

  const float* query = (const float*)d_in[0];
  const float* key_  = (const float*)d_in[1];
  const float* value = (const float*)d_in[2];
  const float* Wq    = (const float*)d_in[3];
  const float* bq    = (const float*)d_in[4];
  const float* Wk    = (const float*)d_in[5];
  const float* bk    = (const float*)d_in[6];
  const float* Wv    = (const float*)d_in[7];
  const float* bv    = (const float*)d_in[8];
  float* out = (float*)d_out;

  const int N = 4096, M = 4096, E = 1024, D = 1024;

  // Workspace layout (~150 MB).
  unsigned short* Xq  = (unsigned short*)d_ws;               // N*D bf16 inputs
  unsigned short* Xk  = Xq  + (size_t)N * D;
  unsigned short* Xv  = Xk  + (size_t)M * D;
  unsigned short* Wqb = Xv  + (size_t)M * D;                 // E*D bf16 weights
  unsigned short* Wkb = Wqb + (size_t)E * D;
  unsigned short* Wvb = Wkb + (size_t)E * D;
  unsigned short* Qb  = Wvb + (size_t)E * D;                 // N*E bf16 projections
  unsigned short* Kb  = Qb  + (size_t)N * E;
  unsigned short* Vb  = Kb  + (size_t)M * E;
  float*          S   = (float*)(Vb + (size_t)M * E);        // N*M f32 scores
  unsigned short* P   = (unsigned short*)(S + (size_t)N * M);// N*M bf16 probs

  dim3 blk(256);

  // Stage 0: one-shot fp32 -> bf16 conversions.
  f32_to_bf16_kernel<<<(N * D / 4 + 255) / 256, blk, 0, stream>>>(query, Xq, N * D);
  f32_to_bf16_kernel<<<(M * D / 4 + 255) / 256, blk, 0, stream>>>(key_,  Xk, M * D);
  f32_to_bf16_kernel<<<(M * D / 4 + 255) / 256, blk, 0, stream>>>(value, Xv, M * D);
  f32_to_bf16_kernel<<<(E * D / 4 + 255) / 256, blk, 0, stream>>>(Wq, Wqb, E * D);
  f32_to_bf16_kernel<<<(E * D / 4 + 255) / 256, blk, 0, stream>>>(Wk, Wkb, E * D);
  f32_to_bf16_kernel<<<(E * D / 4 + 255) / 256, blk, 0, stream>>>(Wv, Wvb, E * D);

  // Projections: bf16(X @ W^T + b) via TDM-staged NT GEMM.
  gemm_nt_bf16_kernel<true><<<dim3(E / BN, N / BM), blk, 0, stream>>>(
      Xq, Wqb, bq, nullptr, Qb, N, E, D);
  gemm_nt_bf16_kernel<true><<<dim3(E / BN, M / BM), blk, 0, stream>>>(
      Xk, Wkb, bk, nullptr, Kb, M, E, D);
  gemm_nt_bf16_kernel<true><<<dim3(E / BN, M / BM), blk, 0, stream>>>(
      Xv, Wvb, bv, nullptr, Vb, M, E, D);

  // S = Qb @ Kb^T (f32, L2-resident).
  gemm_nt_bf16_kernel<false><<<dim3(M / BN, N / BM), blk, 0, stream>>>(
      Qb, Kb, nullptr, S, nullptr, N, M, E);

  // P = softmax_rows(S) in bf16.
  softmax_rows_kernel<<<dim3(N), blk, 0, stream>>>(S, P, M);

  // out = P @ Vb (f32).
  gemm_nn_bf16_kernel<<<dim3(E / BN, N / BM), blk, 0, stream>>>(P, Vb, out, N, E, M);
}
